// MSTAttention_35656818491420
// MI455X (gfx1250) — matrix operand order repaired and checked by reading, not compiled, not run on hardware
//
#include <hip/hip_runtime.h>
#include <hip/hip_bf16.h>

// ---------------------------------------------------------------------------
// MI455X (gfx1250) multi-modality attention, bf16 WMMA pipeline.
//   B=2, S=2048, D=2048, H=16 heads x 128, KV=4 heads (GQA rep=4), M=2 modalities
// GEMMs: v_wmma_f32_16x16x32_bf16; weight tiles staged block-wide in LDS via
// async global->LDS loads (double-buffered, s_wait_asynccnt) when available.
// ---------------------------------------------------------------------------

typedef __attribute__((ext_vector_type(16))) __bf16 bf16x16;
typedef __attribute__((ext_vector_type(8)))  float  f32x8;
typedef __attribute__((ext_vector_type(4)))  unsigned int u32x4;
typedef __attribute__((ext_vector_type(4)))  int i32x4;

typedef __attribute__((address_space(1))) i32x4 gi32x4;  // global v4i
typedef __attribute__((address_space(3))) i32x4 si32x4;  // LDS v4i

union Frag { u32x4 u[2]; bf16x16 v; };

static constexpr int Bc  = 2;
static constexpr int Sc  = 2048;
static constexpr int Dc  = 2048;
static constexpr int Hc  = 16;
static constexpr int KVc = 4;
static constexpr int HDc = 128;
static constexpr int E_Q  = Hc * HDc;   // 2048
static constexpr int E_KV = KVc * HDc;  // 512
static constexpr int LPAD = 72;         // padded LDS row stride (halfwords)

__device__ __forceinline__ unsigned short f2bf(float f) {
  union { float f; unsigned u; } v; v.f = f;
  unsigned r = v.u + 0x7FFFu + ((v.u >> 16) & 1u);   // round-to-nearest-even
  return (unsigned short)(r >> 16);
}

__device__ __forceinline__ f32x8 wmma_bf16(const bf16x16& a, const bf16x16& b, const f32x8& c) {
  // D = A(16x32 bf16) * B(32x16 bf16) + C(16x16 f32)
  return __builtin_amdgcn_wmma_f32_16x16x32_bf16(false, a, false, b, (short)0, c, false, false);
}

__device__ __forceinline__ float rmax16(float v) {
  v = fmaxf(v, __shfl_xor(v, 1, 32));
  v = fmaxf(v, __shfl_xor(v, 2, 32));
  v = fmaxf(v, __shfl_xor(v, 4, 32));
  v = fmaxf(v, __shfl_xor(v, 8, 32));
  return v;
}
__device__ __forceinline__ float rsum16(float v) {
  v += __shfl_xor(v, 1, 32);
  v += __shfl_xor(v, 2, 32);
  v += __shfl_xor(v, 4, 32);
  v += __shfl_xor(v, 8, 32);
  return v;
}

// 16-byte global -> LDS copy; async (ASYNCcnt-tracked) when toolchain exposes it.
__device__ __forceinline__ void copy_b128_to_lds(const unsigned short* g, unsigned short* l) {
#if __has_builtin(__builtin_amdgcn_global_load_async_to_lds_b128)
  __builtin_amdgcn_global_load_async_to_lds_b128((gi32x4*)g, (si32x4*)l, 0, 0);
#else
  *(u32x4*)l = *(const u32x4*)g;
#endif
}

// Stage one 64-row x 32-col bf16 weight tile for both modalities into LDS.
// W has 2048 rows of E columns; 128 threads move 2*32*64 bf16 = 8KB (4 x b128 each).
template <int E>
__device__ __forceinline__ void stage_w(const unsigned short* __restrict__ wb, int k0,
                                        int colbase, unsigned short (*lb)[32][LPAD], int tid) {
#pragma unroll
  for (int ii = 0; ii < 4; ++ii) {
    const int c = ii * 128 + tid;             // 0..511
    const int m = c >> 8, rem = c & 255;
    const int row = rem >> 2, seg = rem & 3;  // 64 rows x 4 x 16B segments
    copy_b128_to_lds(wb + (size_t)m * Dc * E + (size_t)(k0 + row) * E + colbase + seg * 8,
                     &lb[m][row][seg * 8]);
  }
}

// -------------------------- prep: f32 -> bf16 ------------------------------
__global__ void k_f2bf(const float* __restrict__ in, unsigned short* __restrict__ out, int n) {
  for (int i = blockIdx.x * blockDim.x + threadIdx.x; i < n; i += gridDim.x * blockDim.x)
    out[i] = f2bf(in[i]);
}

// Wk/Wv: (M, M, D, E) -> sum over axis 1 -> bf16 (M, D, E)
__global__ void k_sum2_f2bf(const float* __restrict__ in, unsigned short* __restrict__ out, int n, int halfsz) {
  for (int i = blockIdx.x * blockDim.x + threadIdx.x; i < n; i += gridDim.x * blockDim.x) {
    int m = i / halfsz, rest = i - m * halfsz;
    size_t base = (size_t)m * 2 * halfsz + rest;
    out[i] = f2bf(in[base] + in[base + halfsz]);
  }
}

// ------------------- QKV projection (both modalities + select + RoPE) ------
// Block = 4 waves sharing one 64-column weight tile (staged in LDS), each wave
// owns a different 16-token tile.
// MODE 0: Q (E=2048) -> qb[t][e] bf16 (RoPE)
// MODE 1: K (E=512)  -> kt[b][kvh][d][s] bf16 transposed (RoPE)
// MODE 2: V (E=512)  -> vbuf[b][kvh][s][d] bf16
template <int MODE>
__global__ __launch_bounds__(128) void k_qkv_proj(
    const unsigned short* __restrict__ xb,   // [B*S][D] bf16
    const unsigned short* __restrict__ wb,   // [2][D][E] bf16
    const float* __restrict__ freq,          // [S][64][2][2] f32
    const int* __restrict__ mid,             // [B*S]
    unsigned short* __restrict__ out) {
  constexpr int E = (MODE == 0) ? E_Q : E_KV;
  constexpr int NCG = E / 64;
  __shared__ __align__(16) unsigned short lbuf[2][2][32][LPAD];  // dbl-buffered B tiles

  const int tid   = threadIdx.x;
  const int lane  = tid & 31;
  const int wslot = tid >> 5;
  const int tgroup = blockIdx.x / NCG;
  const int cg     = blockIdx.x % NCG;
  const int ttile  = tgroup * 4 + wslot;
  const int tbase  = ttile * 16;
  const int r = lane & 15, half = lane >> 4;
  const int colbase = cg * 64;

  f32x8 acc[2][4];
#pragma unroll
  for (int m = 0; m < 2; ++m)
#pragma unroll
    for (int nt = 0; nt < 4; ++nt) acc[m][nt] = (f32x8){0,0,0,0,0,0,0,0};

  const unsigned short* arow = xb + (size_t)(tbase + r) * Dc + half * 8;
  const int krow = r + half * 16;  // B-matrix K row within 32-tile

  stage_w<E>(wb, 0, colbase, lbuf[0], tid);
  int bi = 0;
  for (int k0 = 0; k0 < Dc; k0 += 32, bi ^= 1) {
    asm volatile("s_wait_asynccnt 0" ::: "memory");
    __syncthreads();
    if (k0 + 32 < Dc) stage_w<E>(wb, k0 + 32, colbase, lbuf[bi ^ 1], tid);
    Frag a;
    a.u[0] = *(const u32x4*)(arow + k0);
    a.u[1] = *(const u32x4*)(arow + k0 + 16);
    __builtin_prefetch(arow + k0 + 64, 0, 1);          // global_prefetch
#pragma unroll
    for (int m = 0; m < 2; ++m) {
      const unsigned short* lrow = &lbuf[bi][m][krow][0];
#pragma unroll
      for (int nt = 0; nt < 4; ++nt) {
        Frag bf;
#pragma unroll
        for (int s = 0; s < 16; ++s)
          ((unsigned short*)&bf)[s] = lbuf[bi][m][(s & 7) + half * 8 + (s >> 3) * 16][nt * 16 + r];
        acc[m][nt] = wmma_bf16(a.v, bf.v, acc[m][nt]);
      }
    }
  }

#pragma unroll
  for (int nt = 0; nt < 4; ++nt) {
    const int col = colbase + nt * 16 + r;
#pragma unroll
    for (int i = 0; i < 8; ++i) {
      const int t = tbase + half * 8 + i;              // flat token id
      const int m = mid[t];
      float v = (m == 0) ? acc[0][nt][i] : acc[1][nt][i];
      const int b = t >> 11, s = t & (Sc - 1);
      if (MODE < 2) {  // RoPE on Q and K: pairs live on adjacent lanes
        const int d  = col & (HDc - 1);
        const int fi = d >> 1;
        const float* f = freq + (size_t)s * 256 + fi * 4;
        const float c = f[0];     // cos
        const float sn = f[2];    // sin
        const float p = __shfl_xor(v, 1, 32);
        v = (lane & 1) ? (v * c - p * sn) : (v * c + p * sn);
      }
      if (MODE == 0) {
        out[(size_t)t * E_Q + col] = f2bf(v);
      } else if (MODE == 1) {
        const int kvh = col >> 7, d = col & (HDc - 1);
        out[(((size_t)(b * KVc + kvh) * HDc + d) << 11) + s] = f2bf(v);
      } else {
        const int kvh = col >> 7, d = col & (HDc - 1);
        out[(((size_t)(b * KVc + kvh) * Sc + s) << 7) + d] = f2bf(v);
      }
    }
  }
}

// --------------------------- flash attention -------------------------------
__global__ __launch_bounds__(128) void k_attn(
    const unsigned short* __restrict__ qb,   // [B*S][2048] bf16 (RoPEd)
    const unsigned short* __restrict__ ktb,  // [B][4][128][S] bf16 (RoPEd, transposed)
    const unsigned short* __restrict__ vb,   // [B][4][S][128] bf16
    unsigned short* __restrict__ ab) {       // [B*S][2048] bf16
  __shared__ __align__(16) unsigned short pbuf[4][16][32];
  const int lane  = threadIdx.x & 31;
  const int wslot = threadIdx.x >> 5;
  const int wid   = blockIdx.x * 4 + wslot;
  const int qtile = wid & 127;
  const int h     = (wid >> 7) & (Hc - 1);
  const int b     = wid >> 11;
  const int kvh   = h >> 2;                  // rep = 4
  const int r = lane & 15, half = lane >> 4;

  // Preload full Q tile (16 queries x 128 dims) as 4 A-frags
  Frag qf[4];
  const unsigned short* qrow = qb + (size_t)(b * Sc + qtile * 16 + r) * E_Q + h * HDc + half * 8;
#pragma unroll
  for (int ks = 0; ks < 4; ++ks) {
    qf[ks].u[0] = *(const u32x4*)(qrow + ks * 32);
    qf[ks].u[1] = *(const u32x4*)(qrow + ks * 32 + 16);
  }
  const unsigned short* ktbase = ktb + (size_t)(b * KVc + kvh) * HDc * Sc;  // [d][s]
  const unsigned short* vbase  = vb  + (size_t)(b * KVc + kvh) * Sc * HDc;  // [s][d]

  f32x8 o[8];
  float mrow[8], lrow[8];
#pragma unroll
  for (int j = 0; j < 8; ++j) o[j] = (f32x8){0,0,0,0,0,0,0,0};
#pragma unroll
  for (int i = 0; i < 8; ++i) { mrow[i] = -1e30f; lrow[i] = 0.0f; }
  const float scale = 0.08838834764831845f;  // 1/sqrt(128)

  for (int kb = 0; kb < Sc; kb += 32) {
    // ---- scores: Q(16x128) x K^T(128x32) as two 16x16 WMMA C-tiles ----
    f32x8 sc[2];
#pragma unroll
    for (int nt = 0; nt < 2; ++nt) {
      f32x8 a = (f32x8){0,0,0,0,0,0,0,0};
      const int keyb = kb + nt * 16;
#pragma unroll
      for (int ks = 0; ks < 4; ++ks) {
        const int krow = ks * 32 + r + half * 16;      // dim index (B row)
        const unsigned short* kp = ktbase + (size_t)krow * Sc + keyb;
        Frag bf;
#pragma unroll
        for (int s = 0; s < 16; ++s)
          ((unsigned short*)&bf)[s] = ktbase[(size_t)(ks * 32 + (s & 7) + half * 8 + (s >> 3) * 16) * Sc + keyb + r];
        a = wmma_bf16(qf[ks].v, bf.v, a);
      }
      sc[nt] = a;
    }
    // ---- online softmax over this 32-key block ----
#pragma unroll
    for (int i = 0; i < 8; ++i) {
      const float s0 = sc[0][i] * scale, s1 = sc[1][i] * scale;
      const float mx = rmax16(fmaxf(s0, s1));
      const float mnew = fmaxf(mrow[i], mx);
      const float alpha = __expf(mrow[i] - mnew);
      const float p0 = __expf(s0 - mnew);
      const float p1 = __expf(s1 - mnew);
      lrow[i] = lrow[i] * alpha + rsum16(p0 + p1);
      mrow[i] = mnew;
#pragma unroll
      for (int j = 0; j < 8; ++j) o[j][i] *= alpha;
      pbuf[wslot][half * 8 + i][r]      = f2bf(p0);
      pbuf[wslot][half * 8 + i][16 + r] = f2bf(p1);
    }
    asm volatile("s_wait_dscnt 0" ::: "memory");       // C-layout -> A-layout via LDS
    Frag pa;
    pa.u[0] = *(const u32x4*)&pbuf[wslot][r][half * 8];
    pa.u[1] = *(const u32x4*)&pbuf[wslot][r][16 + half * 8];
    asm volatile("s_wait_dscnt 0" ::: "memory");
    // ---- O += P(16x32) x V(32x128) ----
    const int krow = kb + r + half * 16;               // key index (B row)
    const unsigned short* vp = vbase + (size_t)krow * HDc;
#pragma unroll
    for (int j = 0; j < 8; ++j) {
      Frag bf;
#pragma unroll
      for (int s = 0; s < 16; ++s)
        ((unsigned short*)&bf)[s] = vbase[(size_t)(kb + (s & 7) + half * 8 + (s >> 3) * 16) * HDc + j * 16 + r];
      o[j] = wmma_bf16(pa.v, bf.v, o[j]);
    }
  }
  // ---- normalize + store bf16 [b][s][h*128+d] ----
#pragma unroll
  for (int j = 0; j < 8; ++j)
#pragma unroll
    for (int i = 0; i < 8; ++i) {
      const float v = o[j][i] / lrow[i];
      const int s = qtile * 16 + half * 8 + i;
      ab[(size_t)(b * Sc + s) * E_Q + h * HDc + j * 16 + r] = f2bf(v);
    }
}

// ---------------- output projection (both modalities + select) -------------
__global__ __launch_bounds__(128) void k_oproj(
    const unsigned short* __restrict__ ab,   // [B*S][2048] bf16
    const unsigned short* __restrict__ wob,  // [2][2048][2048] bf16
    const int* __restrict__ mid,
    float* __restrict__ out) {               // [B*S][2048] f32
  constexpr int NCG = Dc / 64;
  __shared__ __align__(16) unsigned short lbuf[2][2][32][LPAD];

  const int tid   = threadIdx.x;
  const int lane  = tid & 31;
  const int wslot = tid >> 5;
  const int tgroup = blockIdx.x / NCG;
  const int cg     = blockIdx.x % NCG;
  const int ttile  = tgroup * 4 + wslot;
  const int tbase  = ttile * 16;
  const int r = lane & 15, half = lane >> 4;
  const int colbase = cg * 64;

  f32x8 acc[2][4];
#pragma unroll
  for (int m = 0; m < 2; ++m)
#pragma unroll
    for (int nt = 0; nt < 4; ++nt) acc[m][nt] = (f32x8){0,0,0,0,0,0,0,0};

  const unsigned short* arow = ab + (size_t)(tbase + r) * E_Q + half * 8;
  const int krow = r + half * 16;

  stage_w<Dc>(wob, 0, colbase, lbuf[0], tid);
  int bi = 0;
  for (int k0 = 0; k0 < E_Q; k0 += 32, bi ^= 1) {
    asm volatile("s_wait_asynccnt 0" ::: "memory");
    __syncthreads();
    if (k0 + 32 < E_Q) stage_w<Dc>(wob, k0 + 32, colbase, lbuf[bi ^ 1], tid);
    Frag a;
    a.u[0] = *(const u32x4*)(arow + k0);
    a.u[1] = *(const u32x4*)(arow + k0 + 16);
    __builtin_prefetch(arow + k0 + 64, 0, 1);
#pragma unroll
    for (int m = 0; m < 2; ++m) {
      const unsigned short* lrow = &lbuf[bi][m][krow][0];
#pragma unroll
      for (int nt = 0; nt < 4; ++nt) {
        Frag bf;
#pragma unroll
        for (int s = 0; s < 16; ++s)
          ((unsigned short*)&bf)[s] = lbuf[bi][m][(s & 7) + half * 8 + (s >> 3) * 16][nt * 16 + r];
        acc[m][nt] = wmma_bf16(a.v, bf.v, acc[m][nt]);
      }
    }
  }
#pragma unroll
  for (int nt = 0; nt < 4; ++nt) {
    const int col = colbase + nt * 16 + r;
#pragma unroll
    for (int i = 0; i < 8; ++i) {
      const int t = tbase + half * 8 + i;
      const int m = mid[t];
      out[(size_t)t * Dc + col] = (m == 0) ? acc[0][nt][i] : acc[1][nt][i];
    }
  }
}

// ---------------------------------------------------------------------------
extern "C" void kernel_launch(void* const* d_in, const int* in_sizes, int n_in,
                              void* d_out, int out_size, void* d_ws, size_t ws_size,
                              hipStream_t stream) {
  const float* x    = (const float*)d_in[0];
  const float* freq = (const float*)d_in[1];
  const float* Wq   = (const float*)d_in[2];
  const float* Wk   = (const float*)d_in[3];
  const float* Wv   = (const float*)d_in[4];
  const float* Wo   = (const float*)d_in[5];
  const int*   mid  = (const int*)d_in[6];
  float* out = (float*)d_out;

  char* ws = (char*)d_ws;
  size_t off = 0;
  auto alloc = [&](size_t bytes) {
    void* p = ws + off;
    off += (bytes + 255) & ~(size_t)255;
    return p;
  };
  const size_t nTok = (size_t)Bc * Sc;                         // 4096
  unsigned short* xb  = (unsigned short*)alloc(nTok * Dc * 2);         // x bf16
  unsigned short* wqb = (unsigned short*)alloc((size_t)2 * Dc * E_Q * 2);
  unsigned short* wkb = (unsigned short*)alloc((size_t)2 * Dc * E_KV * 2);
  unsigned short* wvb = (unsigned short*)alloc((size_t)2 * Dc * E_KV * 2);
  unsigned short* wob = (unsigned short*)alloc((size_t)2 * E_Q * Dc * 2);
  unsigned short* qbf = (unsigned short*)alloc(nTok * E_Q * 2);        // Q (RoPE)
  unsigned short* ktb = (unsigned short*)alloc((size_t)Bc * KVc * HDc * Sc * 2); // K^T
  unsigned short* vbf = (unsigned short*)alloc((size_t)Bc * KVc * Sc * HDc * 2); // V
  unsigned short* abf = (unsigned short*)alloc(nTok * E_Q * 2);        // attn out

  // --- prep: converts & Wk/Wv axis-1 sums ---
  {
    int n = (int)(nTok * Dc);
    k_f2bf<<<(n + 1023) / 1024, 256, 0, stream>>>(x, xb, n);
  }
  {
    int n = 2 * Dc * E_Q;
    k_f2bf<<<(n + 1023) / 1024, 256, 0, stream>>>(Wq, wqb, n);
    k_f2bf<<<(n + 1023) / 1024, 256, 0, stream>>>(Wo, wob, n);
  }
  {
    int n = 2 * Dc * E_KV, halfsz = Dc * E_KV;
    k_sum2_f2bf<<<(n + 1023) / 1024, 256, 0, stream>>>(Wk, wkb, n, halfsz);
    k_sum2_f2bf<<<(n + 1023) / 1024, 256, 0, stream>>>(Wv, wvb, n, halfsz);
  }

  // --- QKV projections: block = 4 token-tiles sharing one staged weight tile ---
  {
    int blkQ  = ((int)(nTok / 16) / 4) * (E_Q / 64);   // 64*32 = 2048
    int blkKV = ((int)(nTok / 16) / 4) * (E_KV / 64);  // 64*8  = 512
    k_qkv_proj<0><<<blkQ, 128, 0, stream>>>(xb, wqb, freq, mid, qbf);
    k_qkv_proj<1><<<blkKV, 128, 0, stream>>>(xb, wkb, freq, mid, ktb);
    k_qkv_proj<2><<<blkKV, 128, 0, stream>>>(xb, wvb, freq, mid, vbf);
  }

  // --- flash attention: one wave per (b, h, 16-query tile) ---
  {
    int waves = Bc * Hc * (Sc / 16);  // 4096
    k_attn<<<waves / 4, 128, 0, stream>>>(qbf, ktb, vbf, abf);
  }

  // --- output projection ---
  {
    int blk = ((int)(nTok / 16) / 4) * (Dc / 64);  // 2048
    k_oproj<<<blk, 128, 0, stream>>>(abf, wob, mid, out);
  }
  (void)in_sizes; (void)n_in; (void)out_size; (void)ws_size;
}
